// GCNJointRepresentation_71622874628514
// MI455X (gfx1250) — matrix operand order repaired and checked by reading, not compiled, not run on hardware
//
#include <hip/hip_runtime.h>
#include <hip/hip_bf16.h>

typedef __attribute__((ext_vector_type(16))) _Float16 v16h;
typedef __attribute__((ext_vector_type(8)))  _Float16 v8h;
typedef __attribute__((ext_vector_type(8)))  float    v8f;

#define N_NODES 50000
#define N_EDGES 1600000
#define N_TRAIN 1000000
#define F1 128
#define F2 64
#define NCLS 5

// ---------------- degree / normalization ----------------

__global__ void k_deg(const int* __restrict__ dst, float* __restrict__ deg) {
    int e = blockIdx.x * blockDim.x + threadIdx.x;
    if (e < N_EDGES) atomicAdd(&deg[dst[e]], 1.0f);
}

__global__ void k_dinv(float* __restrict__ deg) {
    int i = blockIdx.x * blockDim.x + threadIdx.x;
    if (i < N_NODES) deg[i] = rsqrtf(deg[i] + 1.0f);   // in-place: deg -> d^{-1/2}
}

// ---------------- layer 1 (rank-1 factorized aggregation) ----------------
// h1[i][j] = x[i]*W1[j]  =>  agg1[d][j] = W1[j] * sum_e x[src]*dinv[src]*dinv[d]
__global__ void k_scatter1(const int* __restrict__ src, const int* __restrict__ dst,
                           const float* __restrict__ x, const float* __restrict__ dinv,
                           float* __restrict__ tsum) {
    int e = blockIdx.x * blockDim.x + threadIdx.x;
    if (e < N_EDGES) {
        int s = src[e], d = dst[e];
        atomicAdd(&tsum[d], x[s] * dinv[s] * dinv[d]);
    }
}

// z1 = relu((tsum[i] + x[i]*dinv[i]^2) * W1[j] + b1[j]), stored fp16 for WMMA
__global__ void k_z1(const float* __restrict__ x, const float* __restrict__ dinv,
                     const float* __restrict__ tsum, const float* __restrict__ W1,
                     const float* __restrict__ b1, _Float16* __restrict__ z1h) {
    int tid = blockIdx.x * blockDim.x + threadIdx.x;
    if (tid < N_NODES * F1) {
        int i = tid >> 7, j = tid & 127;
        float di = dinv[i];
        float v = (tsum[i] + x[i] * di * di) * W1[j] + b1[j];
        z1h[tid] = (_Float16)fmaxf(v, 0.0f);
    }
}

// ---------------- B-operand pre-packing into WMMA per-lane register images ----
// Fragment (nt, kk, lane): 16 contiguous halves = B[K = kk*32 + (lane<16?0:8) + (i&7) + 16*(i>>3)]
//                                                  [N = nt*16 + lane%16]
// One 32-byte aligned block per lane -> two global_load_b128 in the GEMM.

__global__ void k_packW2(const float* __restrict__ W2, _Float16* __restrict__ W2p) {
    int tid = blockIdx.x * blockDim.x + threadIdx.x;   // 4 nt * 4 kk * 32 lane * 16 i
    if (tid < 4 * 4 * 32 * 16) {
        int i    = tid & 15;
        int lane = (tid >> 4) & 31;
        int kk   = (tid >> 9) & 3;
        int nt   = tid >> 11;
        int n = nt * 16 + (lane & 15);
        int k = kk * 32 + ((lane < 16) ? 0 : 8) + (i & 7) + ((i >> 3) * 16);
        W2p[tid] = (_Float16)W2[k * F2 + n];
    }
}

// Wl [64,5] -> packed fragments for 2 k-steps, zero-padded to 16 columns
__global__ void k_packWl(const float* __restrict__ Wl, _Float16* __restrict__ Wlp) {
    int tid = blockIdx.x * blockDim.x + threadIdx.x;   // 2 kk * 32 lane * 16 i
    if (tid < 2 * 32 * 16) {
        int i    = tid & 15;
        int lane = (tid >> 4) & 31;
        int kk   = tid >> 9;
        int n = lane & 15;
        int k = kk * 32 + ((lane < 16) ? 0 : 8) + (i & 7) + ((i >> 3) * 16);
        Wlp[tid] = (n < NCLS) ? (_Float16)Wl[k * NCLS + n] : (_Float16)0.0f;
    }
}

// ---------------- GEMM: h2[N,64] = z1h[N,128] @ W2[128,64] via WMMA ----------------
// One wave per 16x16 output tile; K=128 -> 4 chained v_wmma_f32_16x16x32_f16.
__global__ void k_gemm_wmma(const _Float16* __restrict__ z1h,
                            const _Float16* __restrict__ W2p,
                            float* __restrict__ h2) {
    int lane = threadIdx.x & 31;
    int wave = threadIdx.x >> 5;
    int tile = blockIdx.x * 8 + wave;
    if (tile >= (N_NODES / 16) * (F2 / 16)) return;   // wave-uniform: EXEC stays full
    int mt = tile >> 2, nt = tile & 3;
    int mlo  = lane & 15;
    bool hi  = lane >= 16;
    int row  = mt * 16 + mlo;            // A row (lane % 16)
    int n    = nt * 16 + mlo;            // C column (lane % 16)
    int khalfA = hi ? 8 : 0;             // A: lanes>=16 hold K {8..15, 24..31}

    v8f c = {};
    #pragma unroll
    for (int kk = 0; kk < 4; ++kk) {
        int k0 = kk * 32;
        // A: two aligned 16-byte loads per lane (ISA 16-bit A 16x32 layout)
        const _Float16* ap = z1h + row * F1 + k0 + khalfA;
        v8h alo = *(const v8h*)(ap);          // K = k0+khalf .. +7
        v8h ahi = *(const v8h*)(ap + 16);     // K = k0+16+khalf .. +7
        v16h a;
        #pragma unroll
        for (int i = 0; i < 8; ++i) { a[i] = alo[i]; a[i + 8] = ahi[i]; }
        // B: one 32-byte pre-packed per-lane fragment
        v16h b = *(const v16h*)(W2p + ((nt * 4 + kk) * 32 + lane) * 16);
        c = __builtin_amdgcn_wmma_f32_16x16x32_f16(false, a, false, b, (short)0, c,
                                                   false, false);
    }
    // C/D layout: VGPR r, lane -> M = r + 8*(lane>=16), N = lane%16
    int mstore = mt * 16 + (hi ? 8 : 0);
    #pragma unroll
    for (int r = 0; r < 8; ++r) h2[(mstore + r) * F2 + n] = c[r];
}

// ---------------- layer-2 edge aggregation ----------------

__global__ void k_scatter2(const int* __restrict__ src, const int* __restrict__ dst,
                           const float* __restrict__ dinv, const float* __restrict__ h2,
                           float* __restrict__ agg2) {
    int idx = blockIdx.x * blockDim.x + threadIdx.x;
    if (idx < N_EDGES * F2) {
        int e = idx >> 6, j = idx & 63;
        int s = src[e], d = dst[e];
        atomicAdd(&agg2[d * F2 + j], h2[s * F2 + j] * dinv[s] * dinv[d]);
    }
}

// z2 = relu(agg2 + h2*dinv^2 + b2), in place on agg2
__global__ void k_z2(const float* __restrict__ h2, const float* __restrict__ dinv,
                     const float* __restrict__ b2, float* __restrict__ agg2) {
    int tid = blockIdx.x * blockDim.x + threadIdx.x;
    if (tid < N_NODES * F2) {
        int i = tid >> 6, j = tid & 63;
        float di = dinv[i];
        float v = agg2[tid] + h2[tid] * di * di + b2[j];
        agg2[tid] = fmaxf(v, 0.0f);
    }
}

// ---------------- decode: er = z2[s]*z2[d]; logits = er @ Wl + bl; softmax ----------------
// One wave per 16 train edges; K=64 -> 2 WMMAs; softmax over 5 classes via LDS tile.
__global__ void k_decode(const int* __restrict__ tes, const int* __restrict__ ted,
                         const float* __restrict__ z2, const _Float16* __restrict__ Wlp,
                         const float* __restrict__ bl, float* __restrict__ out) {
    __shared__ float lds[8][16][17];
    int lane = threadIdx.x & 31;
    int wave = threadIdx.x >> 5;
    int tile = blockIdx.x * 8 + wave;
    bool active = tile < (N_TRAIN / 16);   // wave-uniform
    int mlo = lane & 15;
    bool hi = lane >= 16;

    if (active) {
        int e = tile * 16 + mlo;
        int s = tes[e], d = ted[e];
        const float* zs = z2 + s * F2;
        const float* zd = z2 + d * F2;
        int khalf = hi ? 8 : 0;
        v8f c = {};
        #pragma unroll
        for (int kk = 0; kk < 2; ++kk) {
            int k0 = kk * 32;
            v16h a;
            #pragma unroll
            for (int i = 0; i < 8; ++i) {
                int klo = k0 + khalf + i;
                int khi = k0 + 16 + khalf + i;
                a[i]     = (_Float16)(zs[klo] * zd[klo]);   // fused gather-multiply
                a[i + 8] = (_Float16)(zs[khi] * zd[khi]);
            }
            v16h b = *(const v16h*)(Wlp + (kk * 32 + lane) * 16);
            c = __builtin_amdgcn_wmma_f32_16x16x32_f16(false, a, false, b, (short)0, c,
                                                       false, false);
        }
        int mrow = hi ? 8 : 0;
        #pragma unroll
        for (int r = 0; r < 8; ++r) lds[wave][mrow + r][mlo] = c[r];
    }
    __syncthreads();   // reached by every thread (inactive waves too)
    if (active && lane < 16) {
        int e = tile * 16 + lane;
        float v[NCLS];
        float mx = -1e30f;
        #pragma unroll
        for (int cI = 0; cI < NCLS; ++cI) {
            v[cI] = lds[wave][lane][cI] + bl[cI];
            mx = fmaxf(mx, v[cI]);
        }
        float sum = 0.0f;
        #pragma unroll
        for (int cI = 0; cI < NCLS; ++cI) { v[cI] = __expf(v[cI] - mx); sum += v[cI]; }
        float inv = 1.0f / sum;
        #pragma unroll
        for (int cI = 0; cI < NCLS; ++cI) out[e * NCLS + cI] = v[cI] * inv;
    }
}

// ---------------- host ----------------

extern "C" void kernel_launch(void* const* d_in, const int* in_sizes, int n_in,
                              void* d_out, int out_size, void* d_ws, size_t ws_size,
                              hipStream_t stream) {
    (void)in_sizes; (void)n_in; (void)out_size; (void)ws_size;
    const float* x  = (const float*)d_in[0];
    const int*   ei = (const int*)d_in[1];
    const int*  tei = (const int*)d_in[2];
    const float* W1 = (const float*)d_in[3];
    const float* b1 = (const float*)d_in[4];
    const float* W2 = (const float*)d_in[5];
    const float* b2 = (const float*)d_in[6];
    const float* Wl = (const float*)d_in[7];
    const float* bl = (const float*)d_in[8];
    float* out = (float*)d_out;
    const int* src = ei,  *dst = ei  + N_EDGES;
    const int* tes = tei, *ted = tei + N_TRAIN;

    size_t off = 0;
    auto alloc = [&](size_t bytes) -> char* {
        char* p = (char*)d_ws + off;
        off += (bytes + 255) & ~(size_t)255;
        return p;
    };
    float*    dinv = (float*)   alloc((size_t)N_NODES * 4);        // deg -> d^-1/2
    float*    tsum = (float*)   alloc((size_t)N_NODES * 4);
    _Float16* z1h  = (_Float16*)alloc((size_t)N_NODES * F1 * 2);
    _Float16* W2p  = (_Float16*)alloc((size_t)4 * 4 * 32 * 16 * 2);  // packed B frags
    _Float16* Wlp  = (_Float16*)alloc((size_t)2 * 32 * 16 * 2);      // packed Wl frags
    float*    h2   = (float*)   alloc((size_t)N_NODES * F2 * 4);
    float*    agg2 = (float*)   alloc((size_t)N_NODES * F2 * 4);   // becomes z2 in place

    hipMemsetAsync(dinv, 0, (size_t)N_NODES * 4, stream);
    hipMemsetAsync(tsum, 0, (size_t)N_NODES * 4, stream);
    hipMemsetAsync(agg2, 0, (size_t)N_NODES * F2 * 4, stream);

    k_deg     <<<(N_EDGES + 255) / 256, 256, 0, stream>>>(dst, dinv);
    k_dinv    <<<(N_NODES + 255) / 256, 256, 0, stream>>>(dinv);
    k_scatter1<<<(N_EDGES + 255) / 256, 256, 0, stream>>>(src, dst, x, dinv, tsum);
    k_z1      <<<(N_NODES * F1 + 255) / 256, 256, 0, stream>>>(x, dinv, tsum, W1, b1, z1h);
    k_packW2  <<<(4 * 4 * 32 * 16 + 255) / 256, 256, 0, stream>>>(W2, W2p);
    k_packWl  <<<(2 * 32 * 16 + 255) / 256, 256, 0, stream>>>(Wl, Wlp);
    k_gemm_wmma<<<((N_NODES / 16) * (F2 / 16) + 7) / 8, 256, 0, stream>>>(z1h, W2p, h2);
    k_scatter2<<<(N_EDGES * F2 + 255) / 256, 256, 0, stream>>>(src, dst, dinv, h2, agg2);
    k_z2      <<<(N_NODES * F2 + 255) / 256, 256, 0, stream>>>(h2, dinv, b2, agg2);
    k_decode  <<<((N_TRAIN / 16) + 7) / 8, 256, 0, stream>>>(tes, ted, agg2, Wlp, bl, out);
}
